// MuSeGNN_43997644980900
// MI455X (gfx1250) — matrix operand order repaired and checked by reading, not compiled, not run on hardware
//
#include <hip/hip_runtime.h>
#include <hip/hip_bf16.h>

// ---------------- static problem config (matches reference) ----------------
#define Bgr   8
#define Nn    10000
#define NTOT  80000           // Bgr*Nn
#define Ee    320000
#define Dd    128
#define Hh    4
#define Cc    128
#define NQKV  1536            // H*C * 3
#define NCAT  1664            // NQKV + 128 (skip)
#define Ll    3
#define Kk    30
#define HID   512
#define OUTD  2

typedef __attribute__((ext_vector_type(16))) __bf16 v16bf;
typedef __attribute__((ext_vector_type(8)))  __bf16 v8bf;
typedef __attribute__((ext_vector_type(4)))  __bf16 v4bf;
typedef __attribute__((ext_vector_type(8)))  float  v8f;

__device__ __forceinline__ float atomicMaxFloat(float* addr, float value) {
    // works with buffer initialized to -0.0f (bits 0x80000000)
    if (value >= 0.0f)
        return __int_as_float(atomicMax((int*)addr, __float_as_int(value)));
    else
        return __uint_as_float(atomicMin((unsigned int*)addr, __float_as_uint(value)));
}

// ---------------------------------------------------------------------------
// Pack [Wq|Wk|Wv|Wskip] (fp32, row-major K=128 x N=1664) into bf16 in the
// exact B-fragment lane order for v_wmma_f32_16x16x32_bf16:
//   packed[((nt*4 + kt)*32 + lane)*16 + i] = W[kt*32 + 8*(lane>>4) + (i&7) + 16*(i>>3)][nt*16 + (lane&15)]
// so a wave's B fragment is one contiguous 32-byte load per lane.
// ---------------------------------------------------------------------------
__global__ void pack_weights_kernel(const float* __restrict__ Wq, const float* __restrict__ Wk,
                                    const float* __restrict__ Wv, const float* __restrict__ Wsk,
                                    const float* __restrict__ bq, const float* __restrict__ bk,
                                    const float* __restrict__ bv, const float* __restrict__ bsk,
                                    __bf16* __restrict__ Bp, float* __restrict__ biasCat) {
    int t = blockIdx.x * blockDim.x + threadIdx.x;
    if (t < Dd * NCAT) {
        int i    = t & 15;
        int lane = (t >> 4) & 31;
        int kt   = (t >> 9) & 3;
        int nt   = t >> 11;                 // 0..103
        int k = kt * 32 + 8 * (lane >> 4) + (i & 7) + 16 * (i >> 3);
        int n = nt * 16 + (lane & 15);
        float w;
        if      (n < 512)  w = Wq[k * 512 + n];
        else if (n < 1024) w = Wk[k * 512 + (n - 512)];
        else if (n < 1536) w = Wv[k * 512 + (n - 1024)];
        else               w = Wsk[k * 128 + (n - 1536)];
        Bp[t] = (__bf16)w;
    }
    if (t < NCAT) {
        float bb = (t < 512) ? bq[t] : (t < 1024) ? bk[t - 512]
                 : (t < 1536) ? bv[t - 1024] : bsk[t - 1536];
        biasCat[t] = bb;
    }
}

__global__ void cast_bf16_kernel(const float* __restrict__ src, __bf16* __restrict__ dst, int n) {
    int t = blockIdx.x * blockDim.x + threadIdx.x;
    if (t < n) dst[t] = (__bf16)src[t];
}

// ---------------------------------------------------------------------------
// WMMA GEMM: [NTOT,128](bf16) x [128,1664](bf16 packed) -> q|k|v (bf16, +bias)
// and skip (fp32, +bias). One wave per 16x64 output strip.
// ---------------------------------------------------------------------------
__global__ void gemm_qkvs_kernel(const __bf16* __restrict__ hb, const __bf16* __restrict__ Bp,
                                 const float* __restrict__ biasCat,
                                 __bf16* __restrict__ qkv, float* __restrict__ skipb) {
    const int MT = NTOT / 16;     // 5000
    const int NG = NCAT / 64;     // 26
    int wave = blockIdx.x * (blockDim.x >> 5) + (threadIdx.x >> 5);
    int lane = threadIdx.x & 31;
    if (wave >= MT * NG) return;
    int mt = wave / NG;
    int ng = wave % NG;
    int m0 = mt * 16;
    int g  = lane >> 4;           // K-half select per A layout
    int r16 = lane & 15;          // row within tile
    const __bf16* arow = hb + (size_t)(m0 + r16) * Dd + 8 * g;

    v8f acc0 = {}, acc1 = {}, acc2 = {}, acc3 = {};
#pragma unroll
    for (int kt = 0; kt < 4; ++kt) {
        if (kt < 3) __builtin_prefetch(arow + (kt + 1) * 32, 0, 1);
        v8bf lo = *reinterpret_cast<const v8bf*>(arow + kt * 32);
        v8bf hi = *reinterpret_cast<const v8bf*>(arow + kt * 32 + 16);
        v16bf a = __builtin_shufflevector(lo, hi, 0,1,2,3,4,5,6,7,8,9,10,11,12,13,14,15);
        const __bf16* bbase = Bp + (((size_t)(ng * 4 + 0) * 4 + kt) * 32 + lane) * 16;
        // 4 consecutive n-tiles; each B fragment is 32B contiguous per lane
        v16bf b0 = *reinterpret_cast<const v16bf*>(bbase + (size_t)0 * 4 * 32 * 16);
        v16bf b1 = *reinterpret_cast<const v16bf*>(bbase + (size_t)1 * 4 * 32 * 16);
        v16bf b2 = *reinterpret_cast<const v16bf*>(bbase + (size_t)2 * 4 * 32 * 16);
        v16bf b3 = *reinterpret_cast<const v16bf*>(bbase + (size_t)3 * 4 * 32 * 16);
        acc0 = __builtin_amdgcn_wmma_f32_16x16x32_bf16(false, a, false, b0, (short)0, acc0, false, false);
        acc1 = __builtin_amdgcn_wmma_f32_16x16x32_bf16(false, a, false, b1, (short)0, acc1, false, false);
        acc2 = __builtin_amdgcn_wmma_f32_16x16x32_bf16(false, a, false, b2, (short)0, acc2, false, false);
        acc3 = __builtin_amdgcn_wmma_f32_16x16x32_bf16(false, a, false, b3, (short)0, acc3, false, false);
    }

    // C/D layout: VGPR r -> row m0 + r + 8*(lane>>4), col = n0 + (lane&15)
    int colBase = ng * 64 + (lane & 15);
    v8f accs[4] = {acc0, acc1, acc2, acc3};
#pragma unroll
    for (int j = 0; j < 4; ++j) {
        int col = colBase + j * 16;
        float bias = biasCat[col];
#pragma unroll
        for (int r = 0; r < 8; ++r) {
            int row = m0 + r + 8 * g;
            float val = accs[j][r] + bias;
            if (col < NQKV) qkv[(size_t)row * NQKV + col] = (__bf16)val;
            else            skipb[(size_t)row * Dd + (col - NQKV)] = val;
        }
    }
}

// ---------------------------------------------------------------------------
// Per-layer scratch init: agg=0, denom=0, m=-0.0f
// ---------------------------------------------------------------------------
__global__ void init_layer_kernel(float* __restrict__ agg, float* __restrict__ m,
                                  float* __restrict__ denom) {
    int t = blockIdx.x * blockDim.x + threadIdx.x;
    if (t < NTOT * Dd) agg[t] = 0.0f;
    if (t < NTOT * Hh) { m[t] = -0.0f; denom[t] = 0.0f; }
}

// ---------------------------------------------------------------------------
// Edge logits: alpha[e,h] = <q[dst,h,:], k[src,h,:]> / sqrt(C); segment max.
// One wave per edge; lane covers 16 channels (head = lane/8).
// ---------------------------------------------------------------------------
__global__ void edge_logits_kernel(const __bf16* __restrict__ qkv, const int* __restrict__ srcA,
                                   const int* __restrict__ dstA, float* __restrict__ alpha,
                                   float* __restrict__ m) {
    int e = blockIdx.x * (blockDim.x >> 5) + (threadIdx.x >> 5);
    if (e >= Ee) return;
    int lane = threadIdx.x & 31;
    int s = srcA[e], d = dstA[e];
    const __bf16* qp = qkv + (size_t)d * NQKV + lane * 16;          // q at offset 0
    const __bf16* kp = qkv + (size_t)s * NQKV + 512 + lane * 16;    // k at offset 512
    v8bf q0 = *reinterpret_cast<const v8bf*>(qp);
    v8bf q1 = *reinterpret_cast<const v8bf*>(qp + 8);
    v8bf k0 = *reinterpret_cast<const v8bf*>(kp);
    v8bf k1 = *reinterpret_cast<const v8bf*>(kp + 8);
    float sum = 0.0f;
#pragma unroll
    for (int i = 0; i < 8; ++i)
        sum += (float)q0[i] * (float)k0[i] + (float)q1[i] * (float)k1[i];
    sum += __shfl_xor(sum, 1, 32);
    sum += __shfl_xor(sum, 2, 32);
    sum += __shfl_xor(sum, 4, 32);
    if ((lane & 7) == 0) {
        int h = lane >> 3;
        float a = sum * 0.08838834764831845f;  // 1/sqrt(128)
        alpha[(size_t)e * Hh + h] = a;
        atomicMaxFloat(&m[d * Hh + h], a);
    }
}

// e = exp(alpha - m[dst]); denom += e
__global__ void edge_exp_kernel(const int* __restrict__ dstA, float* __restrict__ alpha,
                                const float* __restrict__ m, float* __restrict__ denom) {
    int t = blockIdx.x * blockDim.x + threadIdx.x;
    if (t >= Ee * Hh) return;
    int h = t & (Hh - 1);
    int e = t >> 2;
    int d = dstA[e];
    float v = __expf(alpha[t] - m[d * Hh + h]);
    alpha[t] = v;
    atomicAdd(&denom[d * Hh + h], v);
}

// agg[dst,c] += (1/H) * sum_h (e/denom) * v[src,h,c]; wave per edge, 4 ch/lane
__global__ void edge_agg_kernel(const __bf16* __restrict__ qkv, const int* __restrict__ srcA,
                                const int* __restrict__ dstA, const float* __restrict__ alpha,
                                const float* __restrict__ denom, float* __restrict__ agg) {
    int e = blockIdx.x * (blockDim.x >> 5) + (threadIdx.x >> 5);
    if (e >= Ee) return;
    int lane = threadIdx.x & 31;
    int s = srcA[e], d = dstA[e];
    float a[Hh];
#pragma unroll
    for (int h = 0; h < Hh; ++h)
        a[h] = alpha[(size_t)e * Hh + h] / (denom[d * Hh + h] + 1e-16f) * 0.25f;
    const __bf16* vp = qkv + (size_t)s * NQKV + 1024;  // v at offset 1024
    int c = lane * 4;
    float o0 = 0, o1 = 0, o2 = 0, o3 = 0;
#pragma unroll
    for (int h = 0; h < Hh; ++h) {
        v4bf vv = *reinterpret_cast<const v4bf*>(vp + h * Cc + c);
        o0 += a[h] * (float)vv[0];
        o1 += a[h] * (float)vv[1];
        o2 += a[h] * (float)vv[2];
        o3 += a[h] * (float)vv[3];
    }
    float* dst = agg + (size_t)d * Dd + c;
    atomicAdd(dst + 0, o0);
    atomicAdd(dst + 1, o1);
    atomicAdd(dst + 2, o2);
    atomicAdd(dst + 3, o3);
}

// h = relu(BN(agg + skip)); write fp32 and bf16 copies
__global__ void bn_relu_kernel(const float* __restrict__ agg, const float* __restrict__ skipb,
                               const float* __restrict__ gamma, const float* __restrict__ beta,
                               const float* __restrict__ mean, const float* __restrict__ var,
                               int layer, float* __restrict__ hcur, __bf16* __restrict__ hb) {
    int t = blockIdx.x * blockDim.x + threadIdx.x;
    if (t >= NTOT * Dd) return;
    int c = t & (Dd - 1);
    int pc = layer * Cc + c;
    float x = agg[t] + skipb[t];
    float inv = rsqrtf(var[pc] + 1e-5f);
    float y = (x - mean[pc]) * (gamma[pc] * inv) + beta[pc];
    y = fmaxf(y, 0.0f);
    hcur[t] = y;
    hb[t] = (__bf16)y;
}

// ---------------------------------------------------------------------------
// global_sort_pool: per graph, top-30 rows by channel 127 (desc, stable ties)
// ---------------------------------------------------------------------------
__global__ void sort_pool_kernel(const float* __restrict__ hcur, float* __restrict__ pooled) {
    __shared__ float sval[256];
    __shared__ int   sidx[256];
    __shared__ int   sel[Kk];
    int b = blockIdx.x;
    int tid = threadIdx.x;
    const float* hg = hcur + (size_t)b * Nn * Dd;
    for (int k = 0; k < Kk; ++k) {
        float best = -3.4e38f;
        int bidx = 0x7fffffff;
        for (int n = tid; n < Nn; n += 256) {
            bool taken = false;
            for (int j = 0; j < k; ++j) if (sel[j] == n) { taken = true; break; }
            if (taken) continue;
            float v = hg[(size_t)n * Dd + (Dd - 1)];
            if (v > best || (v == best && n < bidx)) { best = v; bidx = n; }
        }
        sval[tid] = best; sidx[tid] = bidx;
        __syncthreads();
        for (int off = 128; off > 0; off >>= 1) {
            if (tid < off) {
                float ov = sval[tid + off]; int oi = sidx[tid + off];
                if (ov > sval[tid] || (ov == sval[tid] && oi < sidx[tid])) {
                    sval[tid] = ov; sidx[tid] = oi;
                }
            }
            __syncthreads();
        }
        if (tid == 0) sel[k] = sidx[0];
        __syncthreads();
    }
    for (int t = tid; t < Kk * Dd; t += 256) {
        int k = t >> 7, c = t & (Dd - 1);
        pooled[(size_t)b * (Kk * Cc) + t] = hg[(size_t)sel[k] * Dd + c];
    }
}

// ---------------------------------------------------------------------------
// MLP head. Only first K*C+1 = 3841 rows of W1 matter (rest hit zero pads).
// ---------------------------------------------------------------------------
__global__ void mlp1_kernel(const float* __restrict__ pooled, const float* __restrict__ age,
                            const float* __restrict__ W1, const float* __restrict__ b1,
                            float* __restrict__ y1) {
    int wv = blockIdx.x * (blockDim.x >> 5) + (threadIdx.x >> 5);
    if (wv >= Bgr * HID) return;
    int lane = threadIdx.x & 31;
    int b = wv >> 9;
    int j = wv & (HID - 1);
    const float* z = pooled + (size_t)b * (Kk * Cc);
    float sum = 0.0f;
    for (int t = lane; t < Kk * Cc; t += 32)
        sum += z[t] * W1[(size_t)t * HID + j];
    sum += __shfl_xor(sum, 1, 32);
    sum += __shfl_xor(sum, 2, 32);
    sum += __shfl_xor(sum, 4, 32);
    sum += __shfl_xor(sum, 8, 32);
    sum += __shfl_xor(sum, 16, 32);
    if (lane == 0) {
        sum += age[b] * W1[(size_t)(Kk * Cc) * HID + j] + b1[j];
        y1[b * HID + j] = fmaxf(sum, 0.0f);
    }
}

__global__ void mlp2_kernel(const float* __restrict__ y1, const float* __restrict__ W2,
                            const float* __restrict__ b2, float* __restrict__ out) {
    int b = threadIdx.x >> 5;      // one wave per graph (blockDim = 256 = 8 waves)
    int lane = threadIdx.x & 31;
    float s0 = 0.0f, s1 = 0.0f;
    for (int t = lane; t < HID; t += 32) {
        float v = y1[b * HID + t];
        s0 += v * W2[t * OUTD + 0];
        s1 += v * W2[t * OUTD + 1];
    }
#pragma unroll
    for (int msk = 1; msk < 32; msk <<= 1) {
        s0 += __shfl_xor(s0, msk, 32);
        s1 += __shfl_xor(s1, msk, 32);
    }
    if (lane == 0) {
        float z0 = s0 + b2[0], z1 = s1 + b2[1];
        float mx = fmaxf(z0, z1);
        float ls = mx + logf(expf(z0 - mx) + expf(z1 - mx));
        out[b * OUTD + 0] = z0 - ls;
        out[b * OUTD + 1] = z1 - ls;
    }
}

// ---------------------------------------------------------------------------
extern "C" void kernel_launch(void* const* d_in, const int* in_sizes, int n_in,
                              void* d_out, int out_size, void* d_ws, size_t ws_size,
                              hipStream_t stream) {
    const float* x   = (const float*)d_in[0];
    const int*   ei  = (const int*)d_in[1];
    const float* age = (const float*)d_in[2];
    const float* Wq  = (const float*)d_in[3];
    const float* bq  = (const float*)d_in[4];
    const float* Wk  = (const float*)d_in[5];
    const float* bk  = (const float*)d_in[6];
    const float* Wv  = (const float*)d_in[7];
    const float* bv  = (const float*)d_in[8];
    const float* Wsk = (const float*)d_in[9];
    const float* bsk = (const float*)d_in[10];
    const float* bnG = (const float*)d_in[11];
    const float* bnB = (const float*)d_in[12];
    const float* bnM = (const float*)d_in[13];
    const float* bnV = (const float*)d_in[14];
    const float* W1  = (const float*)d_in[15];
    const float* b1  = (const float*)d_in[16];
    const float* W2  = (const float*)d_in[17];
    const float* b2  = (const float*)d_in[18];
    float* out = (float*)d_out;

    const int* srcA = ei;
    const int* dstA = ei + Ee;

    // workspace carve-out (256B aligned slabs)
    char* ws = (char*)d_ws;
    size_t off = 0;
    auto carve = [&](size_t bytes) { size_t o = off; off += (bytes + 255) & ~(size_t)255; return o; };
    __bf16* Bp      = (__bf16*)(ws + carve((size_t)Dd * NCAT * 2));
    float*  biasCat = (float*)(ws + carve((size_t)NCAT * 4));
    __bf16* hb      = (__bf16*)(ws + carve((size_t)NTOT * Dd * 2));
    __bf16* qkv     = (__bf16*)(ws + carve((size_t)NTOT * NQKV * 2));
    float*  skipb   = (float*)(ws + carve((size_t)NTOT * Dd * 4));
    float*  hcur    = (float*)(ws + carve((size_t)NTOT * Dd * 4));
    float*  alpha   = (float*)(ws + carve((size_t)Ee * Hh * 4));
    float*  mbuf    = (float*)(ws + carve((size_t)NTOT * Hh * 4));
    float*  denom   = (float*)(ws + carve((size_t)NTOT * Hh * 4));
    float*  agg     = (float*)(ws + carve((size_t)NTOT * Dd * 4));
    float*  pooled  = (float*)(ws + carve((size_t)Bgr * Kk * Cc * 4));
    float*  y1      = (float*)(ws + carve((size_t)Bgr * HID * 4));

    const int TB = 256;

    // pack weights once per launch (cheap, deterministic)
    pack_weights_kernel<<<(Dd * NCAT + TB - 1) / TB, TB, 0, stream>>>(
        Wq, Wk, Wv, Wsk, bq, bk, bv, bsk, Bp, biasCat);

    // layer-0 input
    cast_bf16_kernel<<<(NTOT * Dd + TB - 1) / TB, TB, 0, stream>>>(x, hb, NTOT * Dd);

    const int gemmWaves  = (NTOT / 16) * (NCAT / 64);       // 130000
    const int gemmBlocks = (gemmWaves + 7) / 8;             // 16250
    const int elemBlocks = (NTOT * Dd + TB - 1) / TB;       // 40000
    const int edgeBlocks = (Ee + 7) / 8;                    // 40000 (wave/edge)
    const int expBlocks  = (Ee * Hh + TB - 1) / TB;         // 5000

    for (int layer = 0; layer < Ll; ++layer) {
        gemm_qkvs_kernel<<<gemmBlocks, TB, 0, stream>>>(hb, Bp, biasCat, qkv, skipb);
        init_layer_kernel<<<elemBlocks, TB, 0, stream>>>(agg, mbuf, denom);
        edge_logits_kernel<<<edgeBlocks, TB, 0, stream>>>(qkv, srcA, dstA, alpha, mbuf);
        edge_exp_kernel<<<expBlocks, TB, 0, stream>>>(dstA, alpha, mbuf, denom);
        edge_agg_kernel<<<edgeBlocks, TB, 0, stream>>>(qkv, srcA, dstA, alpha, denom, agg);
        bn_relu_kernel<<<elemBlocks, TB, 0, stream>>>(agg, skipb, bnG, bnB, bnM, bnV,
                                                      layer, hcur, hb);
    }

    sort_pool_kernel<<<Bgr, TB, 0, stream>>>(hcur, pooled);
    mlp1_kernel<<<(Bgr * HID + 7) / 8, TB, 0, stream>>>(pooled, age, W1, b1, y1);
    mlp2_kernel<<<1, TB, 0, stream>>>(y1, W2, b2, out);
}
